// SelectiveAttentionBlock_11123965296938
// MI455X (gfx1250) — matrix operand order repaired and checked by reading, not compiled, not run on hardware
//
#include <hip/hip_runtime.h>

// ---------------------------------------------------------------------------
// Types for CDNA5 WMMA
// ---------------------------------------------------------------------------
typedef __attribute__((ext_vector_type(16))) __bf16 v16bf;
typedef __attribute__((ext_vector_type(8)))  float  v8f;

#define MODE_BIAS  1
#define MODE_GELU  2
#define MODE_RES   4
#define MODE_BF16  8
#define MODE_TRANS 16

__device__ __forceinline__ unsigned short f2bf(float f) {
    // round-to-nearest-even fp32 -> bf16
    unsigned int u = __float_as_uint(f);
    unsigned int r = u + 0x7FFFu + ((u >> 16) & 1u);
    return (unsigned short)(r >> 16);
}

union FragAB {
    v16bf v;
    unsigned int u[8];
    uint4 q[2];
};

// CDNA5 async memory->LDS copy (16B per lane), tracked by ASYNCcnt.
__device__ __forceinline__ void async_copy_b128(unsigned lds_off, unsigned g_off,
                                                const void* sbase)
{
    asm volatile("global_load_async_to_lds_b128 %0, %1, %2"
                 :
                 : "v"(lds_off), "v"(g_off), "s"(sbase)
                 : "memory");
}

// ---------------------------------------------------------------------------
// Generic bf16 WMMA GEMM (TN form):
//   C[M,N] = A[M,K] (row major bf16) * Bt[N,K]^T (Bt row major bf16, ld = K)
// batched via blockIdx.z with element strides sA/sB/sC.
// Workgroup tile 128x128, 8 waves (wave32), each wave: 32x64 = 2x4 WMMA tiles.
// Both operand tiles are filled with GLOBAL_LOAD_ASYNC_TO_LDS_B128 into
// double-buffered LDS (row stride 40 halfs -> 16B-aligned, conflict-free
// ds_load_b128 fragment reads), software-pipelined one K-step ahead.
// ---------------------------------------------------------------------------
template <int MODE>
__global__ void __launch_bounds__(256)
gemm_wmma_bf16(const unsigned short* __restrict__ A,
               const unsigned short* __restrict__ Bt,
               void* __restrict__ Out,
               const float* __restrict__ bias,
               const float* __restrict__ res,
               int M, int N, int K,
               long long sA, long long sB, long long sC,
               float scale, int transS)
{
    __shared__ __align__(16) unsigned short Alds[2][128 * 40];
    __shared__ __align__(16) unsigned short Blds[2][128 * 40];

    const int t    = threadIdx.x;
    const int lane = t & 31;
    const int wid  = t >> 5;
    const int wM   = wid & 3;        // wave row slab (0..3) -> 32 rows
    const int wN   = wid >> 2;       // wave col slab (0..1) -> 64 cols
    const int lh   = lane >> 4;      // lane half
    const int li   = lane & 15;

    const int tileM = blockIdx.y * 128;
    const int tileN = blockIdx.x * 128;
    const int z     = blockIdx.z;

    const unsigned short* Ab = A  + (size_t)z * (size_t)sA;
    const unsigned short* Bb = Bt + (size_t)z * (size_t)sB;

    // per-thread fill coordinates: 512 16B chunks per 128x32 tile, 2 per thread
    const int r0 = t >> 2,          cc0 = (t & 3) * 8;
    const int r1 = (t + 256) >> 2,  cc1 = ((t + 256) & 3) * 8;

    const unsigned lA0[2] = { (unsigned)(size_t)&Alds[0][r0 * 40 + cc0],
                              (unsigned)(size_t)&Alds[1][r0 * 40 + cc0] };
    const unsigned lA1[2] = { (unsigned)(size_t)&Alds[0][r1 * 40 + cc1],
                              (unsigned)(size_t)&Alds[1][r1 * 40 + cc1] };
    const unsigned lB0[2] = { (unsigned)(size_t)&Blds[0][r0 * 40 + cc0],
                              (unsigned)(size_t)&Blds[1][r0 * 40 + cc0] };
    const unsigned lB1[2] = { (unsigned)(size_t)&Blds[0][r1 * 40 + cc1],
                              (unsigned)(size_t)&Blds[1][r1 * 40 + cc1] };

    auto issue_tile = [&](int buf, int kk) {
        // A rows tileM+r, Bt rows tileN+r, both ld = K, 16B per op
        async_copy_b128(lA0[buf], (unsigned)(((unsigned)(tileM + r0) * (unsigned)K + kk + cc0) * 2u), Ab);
        async_copy_b128(lA1[buf], (unsigned)(((unsigned)(tileM + r1) * (unsigned)K + kk + cc1) * 2u), Ab);
        async_copy_b128(lB0[buf], (unsigned)(((unsigned)(tileN + r0) * (unsigned)K + kk + cc0) * 2u), Bb);
        async_copy_b128(lB1[buf], (unsigned)(((unsigned)(tileN + r1) * (unsigned)K + kk + cc1) * 2u), Bb);
    };

    v8f acc[2][4] = {};

    const int nIter = K >> 5;
    issue_tile(0, 0);

    for (int it = 0; it < nIter; ++it) {
        const int cur = it & 1;
        if (it + 1 < nIter) {
            issue_tile(cur ^ 1, (it + 1) << 5);
            // 4 just-issued may stay outstanding; the 4 for `cur` must be done
            asm volatile("s_wait_asynccnt 0x4" ::: "memory");
        } else {
            asm volatile("s_wait_asynccnt 0x0" ::: "memory");
        }
        __syncthreads();   // all waves' fills of buf[cur] visible

        FragAB fa[2], fb[4];
#pragma unroll
        for (int mi = 0; mi < 2; ++mi) {
            // A 16x32 bf16: lanes0-15 V0-3:K0-7, V4-7:K16-23; lanes16-31 +8
            const unsigned short* ar =
                &Alds[cur][(wM * 32 + mi * 16 + li) * 40 + 8 * lh];
            fa[mi].q[0] = *(const uint4*)(ar);        // K =  8h .. 8h+7
            fa[mi].q[1] = *(const uint4*)(ar + 16);   // K = 16+8h .. 16+8h+7
        }
#pragma unroll
        for (int ni = 0; ni < 4; ++ni) {
            // B 32x16 bf16: lanes0-15 V0-3:K0-7, V4-7:K16-23; lanes16-31 +8
            const unsigned short* br =
                &Blds[cur][(wN * 64 + ni * 16 + li) * 40 + 8 * lh];
            fb[ni].q[0] = *(const uint4*)(br);        // K =  8h .. 8h+7
            fb[ni].q[1] = *(const uint4*)(br + 16);   // K = 16+8h .. 16+8h+7
        }

#pragma unroll
        for (int mi = 0; mi < 2; ++mi)
#pragma unroll
            for (int ni = 0; ni < 4; ++ni)
                acc[mi][ni] = __builtin_amdgcn_wmma_f32_16x16x32_bf16(
                    false, fa[mi].v, false, fb[ni].v,
                    (short)0, acc[mi][ni], false, false);

        __syncthreads();   // reads of buf[cur] done before it gets refilled
    }

    // ---- epilogue ----
#pragma unroll
    for (int mi = 0; mi < 2; ++mi)
#pragma unroll
        for (int ni = 0; ni < 4; ++ni)
#pragma unroll
            for (int v = 0; v < 8; ++v) {
                int gm = tileM + wM * 32 + mi * 16 + 8 * lh + v;
                int gn = tileN + wN * 64 + ni * 16 + li;
                float val = acc[mi][ni][v] * scale;
                if (MODE & MODE_BIAS) val += bias[gn];
                if (MODE & MODE_GELU)
                    val = 0.5f * val * (1.0f + erff(val * 0.70710678118654752f));
                if (MODE & MODE_RES)  val += res[(size_t)gm * (size_t)N + gn];
                if (MODE & MODE_TRANS) {
                    // store bf16 transposed per-batch: out[b][gn][s], rows map b*transS+s
                    int bb = gm / transS;
                    int sr = gm - bb * transS;
                    ((unsigned short*)Out)[((size_t)bb * N + gn) * (size_t)transS + sr] =
                        f2bf(val);
                } else if (MODE & MODE_BF16) {
                    ((unsigned short*)Out)[(size_t)z * (size_t)sC +
                                           (size_t)gm * (size_t)N + gn] = f2bf(val);
                } else {
                    ((float*)Out)[(size_t)z * (size_t)sC +
                                  (size_t)gm * (size_t)N + gn] = val;
                }
            }
}

// ---------------------------------------------------------------------------
// LayerNorm over rows of length 1024 (fp32 in, bf16 out). One 256-thr block/row.
// ---------------------------------------------------------------------------
__global__ void __launch_bounds__(256)
layernorm_bf16(const float* __restrict__ x, const float* __restrict__ g,
               const float* __restrict__ bta, unsigned short* __restrict__ out)
{
    __shared__ float red[256];
    const int t = threadIdx.x;
    const size_t row = blockIdx.x;
    const float* xr = x + row * 1024;
    float v[4];
    float s = 0.f;
#pragma unroll
    for (int j = 0; j < 4; ++j) { v[j] = xr[t + 256 * j]; s += v[j]; }
    red[t] = s; __syncthreads();
    for (int o = 128; o > 0; o >>= 1) { if (t < o) red[t] += red[t + o]; __syncthreads(); }
    float mu = red[0] * (1.f / 1024.f);
    __syncthreads();
    float sq = 0.f;
#pragma unroll
    for (int j = 0; j < 4; ++j) { float d = v[j] - mu; sq += d * d; }
    red[t] = sq; __syncthreads();
    for (int o = 128; o > 0; o >>= 1) { if (t < o) red[t] += red[t + o]; __syncthreads(); }
    float rs = rsqrtf(red[0] * (1.f / 1024.f) + 1e-5f);
#pragma unroll
    for (int j = 0; j < 4; ++j) {
        int c = t + 256 * j;
        out[row * 1024 + c] = f2bf((v[j] - mu) * rs * g[c] + bta[c]);
    }
}

// ---------------------------------------------------------------------------
// Row softmax over 2048 keys with additive per-key bias; fp32 in, bf16 out.
// scores already include the 1/sqrt(D) scale (folded into Q).
// ---------------------------------------------------------------------------
__global__ void __launch_bounds__(256)
softmax_bias_bf16(const float* __restrict__ sc, const float* __restrict__ imp,
                  unsigned short* __restrict__ P)
{
    __shared__ float red[256];
    const int t = threadIdx.x;
    const size_t row = blockIdx.x;        // 0 .. B*S-1
    const size_t bb  = row >> 11;         // row / 2048
    const float* sr = sc + row * 2048;
    const float* ir = imp + bb * 2048;
    float v[8];
    float m = -3.0e38f;
#pragma unroll
    for (int j = 0; j < 8; ++j) {
        int c = t + 256 * j;
        v[j] = sr[c] + ir[c];
        m = fmaxf(m, v[j]);
    }
    red[t] = m; __syncthreads();
    for (int o = 128; o > 0; o >>= 1) { if (t < o) red[t] = fmaxf(red[t], red[t + o]); __syncthreads(); }
    m = red[0]; __syncthreads();
    float s = 0.f;
#pragma unroll
    for (int j = 0; j < 8; ++j) { v[j] = expf(v[j] - m); s += v[j]; }
    red[t] = s; __syncthreads();
    for (int o = 128; o > 0; o >>= 1) { if (t < o) red[t] += red[t + o]; __syncthreads(); }
    float inv = 1.0f / red[0];
#pragma unroll
    for (int j = 0; j < 8; ++j)
        P[row * 2048 + t + 256 * j] = f2bf(v[j] * inv);
}

// ---------------------------------------------------------------------------
// fp32 [R,C] -> bf16 transposed [C,R] (weights; 32x32 LDS tiles, coalesced)
// ---------------------------------------------------------------------------
__global__ void __launch_bounds__(256)
cvt_transpose_bf16(const float* __restrict__ in, unsigned short* __restrict__ out,
                   int R, int C)
{
    __shared__ float tile[32][33];
    const int r0 = blockIdx.y * 32, c0 = blockIdx.x * 32;
    const int tx = threadIdx.x & 31, ty = threadIdx.x >> 5;   // ty 0..7
#pragma unroll
    for (int j = 0; j < 32; j += 8)
        tile[ty + j][tx] = in[(size_t)(r0 + ty + j) * C + c0 + tx];
    __syncthreads();
#pragma unroll
    for (int j = 0; j < 32; j += 8)
        out[(size_t)(c0 + ty + j) * R + r0 + tx] = f2bf(tile[tx][ty + j]);
}

// ---------------------------------------------------------------------------
// Host launcher
// ---------------------------------------------------------------------------
extern "C" void kernel_launch(void* const* d_in, const int* in_sizes, int n_in,
                              void* d_out, int out_size, void* d_ws, size_t ws_size,
                              hipStream_t stream)
{
    (void)in_sizes; (void)n_in; (void)out_size; (void)ws_size;

    const float* tokens = (const float*)d_in[0];
    const float* imp    = (const float*)d_in[1];
    const float* ln1_g  = (const float*)d_in[2];
    const float* ln1_b  = (const float*)d_in[3];
    const float* Wq     = (const float*)d_in[4];
    const float* bq     = (const float*)d_in[5];
    const float* Wk     = (const float*)d_in[6];
    const float* bk     = (const float*)d_in[7];
    const float* Wv     = (const float*)d_in[8];
    const float* bv     = (const float*)d_in[9];
    const float* Wo     = (const float*)d_in[10];
    const float* bo     = (const float*)d_in[11];
    const float* ln2_g  = (const float*)d_in[12];
    const float* ln2_b  = (const float*)d_in[13];
    const float* W1     = (const float*)d_in[14];
    const float* b1     = (const float*)d_in[15];
    const float* W2     = (const float*)d_in[16];
    const float* b2     = (const float*)d_in[17];

    const int Bn = 4, S = 2048, D = 1024, H = 2048;
    const int MS = Bn * S;                     // 8192 tokens

    char* ws = (char*)d_ws;
    size_t off = 0;
    auto take = [&](size_t bytes) -> char* {
        char* p = ws + off;
        off += (bytes + 255) & ~(size_t)255;
        return p;
    };

    // all weight B-operands stored TRANSPOSED (TN gemm)
    unsigned short* wq_t = (unsigned short*)take((size_t)D * D * 2);
    unsigned short* wk_t = (unsigned short*)take((size_t)D * D * 2);
    unsigned short* wv_t = (unsigned short*)take((size_t)D * D * 2);
    unsigned short* wo_t = (unsigned short*)take((size_t)D * D * 2);
    unsigned short* w1_t = (unsigned short*)take((size_t)D * H * 2);  // [H,D]
    unsigned short* w2_t = (unsigned short*)take((size_t)H * D * 2);  // [D,H]
    unsigned short* x_bf = (unsigned short*)take((size_t)MS * D * 2); // ln1 out; reused for ln2 out
    unsigned short* q_bf = (unsigned short*)take((size_t)MS * D * 2);
    unsigned short* k_bf = (unsigned short*)take((size_t)MS * D * 2); // K (natural = Bt of QK^T)
    unsigned short* vt_bf= (unsigned short*)take((size_t)MS * D * 2); // V^T per batch [D,S]
    unsigned short* at_bf= (unsigned short*)take((size_t)MS * D * 2); // attn output
    unsigned short* p_bf = (unsigned short*)take((size_t)Bn * S * S * 2);
    float*          scores = (float*)take((size_t)Bn * S * S * 4);    // 64MB, reused below
    unsigned short* act_bf = (unsigned short*)scores;                 // first 32MB (after softmax)
    float*          x2     = (float*)((char*)scores + (size_t)Bn * S * S * 2); // second 32MB

    // 1) weights -> bf16, transposed
    cvt_transpose_bf16<<<dim3(D / 32, D / 32), 256, 0, stream>>>(Wq, wq_t, D, D);
    cvt_transpose_bf16<<<dim3(D / 32, D / 32), 256, 0, stream>>>(Wk, wk_t, D, D);
    cvt_transpose_bf16<<<dim3(D / 32, D / 32), 256, 0, stream>>>(Wv, wv_t, D, D);
    cvt_transpose_bf16<<<dim3(D / 32, D / 32), 256, 0, stream>>>(Wo, wo_t, D, D);
    cvt_transpose_bf16<<<dim3(H / 32, D / 32), 256, 0, stream>>>(W1, w1_t, D, H);
    cvt_transpose_bf16<<<dim3(D / 32, H / 32), 256, 0, stream>>>(W2, w2_t, H, D);

    // 2) LN1
    layernorm_bf16<<<MS, 256, 0, stream>>>(tokens, ln1_g, ln1_b, x_bf);

    dim3 blk(256);
    dim3 gQKV(D / 128, MS / 128, 1);

    // 3) Q = (x Wq) / sqrt(D) + bq   (scale folded, bias unscaled; bf16)
    gemm_wmma_bf16<MODE_BIAS | MODE_BF16><<<gQKV, blk, 0, stream>>>(
        x_bf, wq_t, q_bf, bq, nullptr, MS, D, D, 0, 0, 0, 0.03125f, 0);
    // 4) K = x Wk + bk (bf16, natural layout == Bt for QK^T)
    gemm_wmma_bf16<MODE_BIAS | MODE_BF16><<<gQKV, blk, 0, stream>>>(
        x_bf, wk_t, k_bf, bk, nullptr, MS, D, D, 0, 0, 0, 1.0f, 0);
    // 5) V^T = (x Wv + bv)^T per batch (bf16, transposed store -> Bt for P@V)
    gemm_wmma_bf16<MODE_BIAS | MODE_TRANS><<<gQKV, blk, 0, stream>>>(
        x_bf, wv_t, vt_bf, bv, nullptr, MS, D, D, 0, 0, 0, 1.0f, S);

    // 6) scores = Q K^T  (batched over B, fp32 out)
    dim3 gSc(S / 128, S / 128, Bn);
    gemm_wmma_bf16<0><<<gSc, blk, 0, stream>>>(
        q_bf, k_bf, scores, nullptr, nullptr, S, S, D,
        (long long)S * D, (long long)S * D, (long long)S * S, 1.0f, 0);

    // 7) P = softmax(scores + importance)  (bf16)
    softmax_bias_bf16<<<MS, 256, 0, stream>>>(scores, imp, p_bf);

    // 8) attn = P V  (batched; Bt = V^T [D,S], fp32 acc, bf16 out)
    dim3 gPV(D / 128, S / 128, Bn);
    gemm_wmma_bf16<MODE_BF16><<<gPV, blk, 0, stream>>>(
        p_bf, vt_bf, at_bf, nullptr, nullptr, S, D, S,
        (long long)S * S, (long long)D * S, (long long)S * D, 1.0f, 0);

    // 9) x2 = tokens + attn Wo + bo  (fp32 out)
    gemm_wmma_bf16<MODE_BIAS | MODE_RES><<<gQKV, blk, 0, stream>>>(
        at_bf, wo_t, x2, bo, tokens, MS, D, D, 0, 0, 0, 1.0f, 0);

    // 10) h = LN2(x2)  (bf16, reuse x_bf)
    layernorm_bf16<<<MS, 256, 0, stream>>>(x2, ln2_g, ln2_b, x_bf);

    // 11) act = gelu(h W1 + b1)  (bf16)
    dim3 gF1(H / 128, MS / 128, 1);
    gemm_wmma_bf16<MODE_BIAS | MODE_GELU | MODE_BF16><<<gF1, blk, 0, stream>>>(
        x_bf, w1_t, act_bf, b1, nullptr, MS, H, D, 0, 0, 0, 1.0f, 0);

    // 12) out = x2 + act W2 + b2  (fp32)
    gemm_wmma_bf16<MODE_BIAS | MODE_RES><<<gQKV, blk, 0, stream>>>(
        act_bf, w2_t, d_out, b2, x2, MS, D, H, 0, 0, 0, 1.0f, 0);
}
